// TernaryLoRAModel_34686155882823
// MI455X (gfx1250) — matrix operand order repaired and checked by reading, not compiled, not run on hardware
//
#include <hip/hip_runtime.h>

// ---------------- problem constants ----------------
#define IN_F   4096
#define OUT_F  11008
#define RANK   16
#define MROWS  256
#define KAUG   (IN_F + 64)        // 4096 real + 16 lora + rowsum + ones + pad
#define NWORDS (IN_F / 16)        // packed int2 words per weight row

// ---------------- tile config ----------------
#define BM 128
#define BN 128
#define BK 64
#define KPAD 72                   // LDS row stride (bf16 elems) = 64 + 4-DWORD TDM pad
#define NSTAGES (KAUG / BK)       // 65

typedef __attribute__((ext_vector_type(16))) __bf16 v16bf;
typedef __attribute__((ext_vector_type(8)))  float  v8f;
typedef unsigned int u32x4 __attribute__((ext_vector_type(4)));
typedef int          i32x4 __attribute__((ext_vector_type(4)));
typedef int          i32x8 __attribute__((ext_vector_type(8)));

#if __has_builtin(__builtin_amdgcn_tensor_load_to_lds) && __has_builtin(__builtin_amdgcn_s_wait_tensorcnt)
#define USE_TDM 1
#else
#define USE_TDM 0
#endif

union Frag {
    uint4 q[2];
    v16bf v;
};

// float -> bf16 (round to nearest even) via integer ops.
__device__ __forceinline__ unsigned short bf16_rne(float f) {
    union { float f; unsigned int u; } v;
    v.f = f;
    unsigned int u = v.u;
    u += 0x7FFFu + ((u >> 16) & 1u);
    return (unsigned short)(u >> 16);
}

// Decode one packed int2 word (16 ternary values) into 8 uints = 16 bf16.
// code 0 -> -1 (0xBF80), 1 -> 0, 2 -> +1 (0x3F80). Exact in bf16.
__device__ __forceinline__ void unpack_word_bf16(unsigned int w, unsigned int* up) {
#pragma unroll
    for (int j = 0; j < 8; ++j) {
        unsigned int c0 = (w >> (4 * j)) & 3u;
        unsigned int c1 = (w >> (4 * j + 2)) & 3u;
        unsigned int lo = (c0 == 0u) ? 0xBF80u : ((c0 == 2u) ? 0x3F80u : 0u);
        unsigned int hi = (c1 == 0u) ? 0xBF80u : ((c1 == 2u) ? 0x3F80u : 0u);
        up[j] = lo | (hi << 16);
    }
}

#if USE_TDM
// Tensor Data Mover: DMA a [BM x BK] bf16 tile (row stride KAUG elems) from
// global into LDS, with LDS padding of 4 DWORDs every 32 DWORDs so the
// landed layout has row stride KPAD = 72 bf16 (bank-conflict-free).
// D# per CDNA5 ISA ch.8: group0 = {count/flags, lds_addr, global_addr, type},
// group1 = {mask/size/pad, tensor dims, tile dims, strides}.
__device__ __forceinline__ void tdm_load_A(const unsigned short* gptr,
                                           unsigned int lds_byte_addr) {
    unsigned long long ga = (unsigned long long)(size_t)gptr;
    u32x4 g0;
    g0[0] = 1u;                                    // count=1 (valid user descriptor)
    g0[1] = lds_byte_addr;                         // lds_addr
    g0[2] = (unsigned int)ga;                      // global_addr[31:0]
    g0[3] = (unsigned int)((ga >> 32) & 0x1FFFFFFull) | (2u << 30); // addr[56:32] | type=2
    i32x8 g1;
    g1[0] = (1 << 16)        // data_size = 2 bytes
          | (1 << 20)        // pad_enable
          | (4 << 22)        // pad_interval: 2^(4+1)=32 DWORDs (= 64 bf16 row)
          | (3 << 25);       // pad_amount: 3+1 = 4 DWORDs (= 8 bf16 -> stride 72)
    g1[1] = (int)((unsigned)KAUG << 16);           // tensor_dim0[15:0] in bits 63:48
    g1[2] = (int)((unsigned)MROWS << 16);          // dim0 hi=0 | tensor_dim1[15:0]
    g1[3] = (int)((unsigned)BK << 16);             // dim1 hi=0 | tile_dim0=64
    g1[4] = BM;                                    // tile_dim1=128 | tile_dim2=0
    g1[5] = KAUG;                                  // tensor_dim0_stride[31:0]
    g1[6] = 0;
    g1[7] = 0;
    i32x4 gz = {0, 0, 0, 0};
#if defined(__clang_major__) && __clang_major__ >= 23
    i32x8 gz8 = {0, 0, 0, 0, 0, 0, 0, 0};
    __builtin_amdgcn_tensor_load_to_lds(g0, g1, gz, gz, gz8, 0);
#else
    __builtin_amdgcn_tensor_load_to_lds(g0, g1, gz, gz, 0);
#endif
}
#endif

// ---------------------------------------------------------------------------
// Prologue 1: build augmented A panel (bf16), one block per row m.
//   Aaug[m, 0:4096]    = bf16(x[m, :])
//   Aaug[m, 4096+r]    = bf16(sum_i x[m,i] * lora_A[r,i])   r < 16
//   Aaug[m, 4112]      = bf16(sum_i x[m,i]);  Aaug[m, 4113] = 1.0; rest 0
// ---------------------------------------------------------------------------
__global__ __launch_bounds__(256) void prep_a_kernel(const float* __restrict__ x,
                                                     const float* __restrict__ lora_A,
                                                     unsigned short* __restrict__ Aaug) {
    __shared__ float red[RANK][257];
    __shared__ float rred[256];
    const int m = blockIdx.x;
    const int t = threadIdx.x;

    float pr[RANK];
#pragma unroll
    for (int r = 0; r < RANK; ++r) pr[r] = 0.0f;
    float rs = 0.0f;

    for (int i = t; i < IN_F; i += 256) {
        const float xv = x[(size_t)m * IN_F + i];
        Aaug[(size_t)m * KAUG + i] = bf16_rne(xv);
        rs += xv;
#pragma unroll
        for (int r = 0; r < RANK; ++r) pr[r] += xv * lora_A[(size_t)r * IN_F + i];
    }
#pragma unroll
    for (int r = 0; r < RANK; ++r) red[r][t] = pr[r];
    rred[t] = rs;
    __syncthreads();

    if (t < RANK) {
        float s = 0.0f;
        for (int j = 0; j < 256; ++j) s += red[t][j];
        Aaug[(size_t)m * KAUG + IN_F + t] = bf16_rne(s);
    }
    if (t == 0) {
        float s = 0.0f;
        for (int j = 0; j < 256; ++j) s += rred[j];
        unsigned short* tail = Aaug + (size_t)m * KAUG + IN_F + RANK;
        tail[0] = bf16_rne(s);        // rowsum column
        tail[1] = 0x3F80u;            // ones column
        for (int j = 2; j < KAUG - IN_F - RANK; ++j) tail[j] = 0u;
    }
}

// ---------------------------------------------------------------------------
// Prologue 2: aux B panel (bf16), Baux[o, 0:64]:
//   [0:16) = 2*lora_B[o,r]/alpha[o];  [16] = mu[o]/alpha[o];
//   [17]   = bias[o]/alpha[o];        rest 0
// ---------------------------------------------------------------------------
__global__ __launch_bounds__(256) void prep_baux_kernel(const float* __restrict__ alpha,
                                                        const float* __restrict__ mu,
                                                        const float* __restrict__ bias,
                                                        const float* __restrict__ lora_B,
                                                        unsigned short* __restrict__ Baux) {
    const int o = blockIdx.x * 256 + threadIdx.x;
    if (o >= OUT_F) return;
    const float inva = 1.0f / alpha[o];
    unsigned short* row = Baux + (size_t)o * 64;
#pragma unroll
    for (int r = 0; r < RANK; ++r)
        row[r] = bf16_rne(2.0f * lora_B[(size_t)o * RANK + r] * inva);
    row[16] = bf16_rne(mu[o] * inva);
    row[17] = bf16_rne(bias[o] * inva);
#pragma unroll
    for (int j = RANK + 2; j < 64; ++j) row[j] = 0u;
}

// ---------------------------------------------------------------------------
// Main GEMM: out[m,o] = alpha[o] * sum_k Aaug[m,k] * Baug[o,k]
// Double-buffered LDS pipeline: A tile via TDM (wave 0 issues, TENSORcnt
// waited before the stage barrier), B tile unpacked int2->bf16 via VALU+DS.
// ---------------------------------------------------------------------------
__global__ __launch_bounds__(256) void ternary_gemm_kernel(
        const unsigned short* __restrict__ Aaug,
        const unsigned short* __restrict__ Baux,
        const int* __restrict__ packed,
        const float* __restrict__ alpha,
        float* __restrict__ out) {
    __shared__ unsigned short As[2][BM][KPAD];
    __shared__ unsigned short Bs[2][BN][KPAD];

    const int tid  = threadIdx.x;
    const int wave = tid >> 5;
    const int lane = tid & 31;
    const int laneHalf = lane >> 4;     // 0: lanes 0-15, 1: lanes 16-31
    const int lr = lane & 15;

    const int wm = wave & 3;            // 0..3 -> M offset wm*32
    const int wn = wave >> 2;           // 0..1 -> N offset wn*64

    const int n0 = blockIdx.x * BN;
    const int m0 = blockIdx.y * BM;

    // staging roles
    const int arow  = tid >> 1;         // 0..127
    const int acol  = (tid & 1) * 32;   // 0 or 32
    const int brow  = tid >> 1;         // 0..127
    const int bhalf = tid & 1;          // which 32-col half

    const unsigned short* Atile0 = Aaug + (size_t)m0 * KAUG;

    v8f acc[2][4] = {};

    // ---- stage issue helpers ----
    auto issue_A = [&](int s, int buf) {
#if USE_TDM
        if (wave == 0)
            tdm_load_A(Atile0 + s * BK,
                       (unsigned int)(size_t)(&As[buf][0][0]));
#else
        const int k0 = s * BK;
        const uint4* gA = (const uint4*)(Aaug + (size_t)(m0 + arow) * KAUG + k0 + acol);
        uint4 a0 = gA[0], a1 = gA[1], a2 = gA[2], a3 = gA[3];
        uint4* sA = (uint4*)(&As[buf][arow][acol]);
        sA[0] = a0; sA[1] = a1; sA[2] = a2; sA[3] = a3;
#endif
    };

    auto issue_B = [&](int s, int buf) {
        const int k0 = s * BK;
        if (k0 < IN_F) {
            const uint2 w2 = *(const uint2*)(packed + (size_t)(n0 + brow) * NWORDS +
                                             (k0 >> 4) + bhalf * 2);
            unsigned int up[16];
            unpack_word_bf16(w2.x, up);
            unpack_word_bf16(w2.y, up + 8);
            uint4* sB = (uint4*)(&Bs[buf][brow][bhalf * 32]);
            sB[0] = make_uint4(up[0],  up[1],  up[2],  up[3]);
            sB[1] = make_uint4(up[4],  up[5],  up[6],  up[7]);
            sB[2] = make_uint4(up[8],  up[9],  up[10], up[11]);
            sB[3] = make_uint4(up[12], up[13], up[14], up[15]);
        } else {
            const uint4* gB = (const uint4*)(Baux + (size_t)(n0 + brow) * 64 + bhalf * 32);
            uint4 b0 = gB[0], b1 = gB[1], b2 = gB[2], b3 = gB[3];
            uint4* sB = (uint4*)(&Bs[buf][brow][bhalf * 32]);
            sB[0] = b0; sB[1] = b1; sB[2] = b2; sB[3] = b3;
        }
    };

    // ---- pipeline prologue ----
    issue_A(0, 0);
    issue_B(0, 0);
#if USE_TDM
    if (wave == 0) __builtin_amdgcn_s_wait_tensorcnt(0);
#endif
    __syncthreads();

    // ---- main pipeline ----
    for (int s = 0; s < NSTAGES; ++s) {
        const int cur = s & 1;
        if (s + 1 < NSTAGES) {
            issue_A(s + 1, cur ^ 1);
            issue_B(s + 1, cur ^ 1);
        }

#pragma unroll
        for (int kk = 0; kk < BK; kk += 32) {
            Frag af[2];
            Frag bfr[4];
#pragma unroll
            for (int ms = 0; ms < 2; ++ms) {
                const int mrow = wm * 32 + ms * 16 + lr;
                // 16-bit A 16x32 layout: lanes 0-15 hold K 0..7 & 16..23,
                // lanes 16-31 hold K 8..15 & 24..31 (VGPR pairs in K order).
                af[ms].q[0] = *(const uint4*)(&As[cur][mrow][kk + laneHalf * 8]);
                af[ms].q[1] = *(const uint4*)(&As[cur][mrow][kk + 16 + laneHalf * 8]);
            }
#pragma unroll
            for (int ns = 0; ns < 4; ++ns) {
                const int ncol = wn * 64 + ns * 16 + lr;
                // 16-bit B 32x16 layout: lanes 0-15 hold K 0..7 & 16..23,
                // lanes 16-31 hold K 8..15 & 24..31 (matches A).
                bfr[ns].q[0] = *(const uint4*)(&Bs[cur][ncol][kk + laneHalf * 8]);
                bfr[ns].q[1] = *(const uint4*)(&Bs[cur][ncol][kk + 16 + laneHalf * 8]);
            }
#pragma unroll
            for (int ms = 0; ms < 2; ++ms)
#pragma unroll
                for (int ns = 0; ns < 4; ++ns)
                    acc[ms][ns] = __builtin_amdgcn_wmma_f32_16x16x32_bf16(
                        false, af[ms].v, false, bfr[ns].v,
                        (short)0, acc[ms][ns], false, false);
        }

#if USE_TDM
        if (wave == 0) __builtin_amdgcn_s_wait_tensorcnt(0);
#endif
        __syncthreads();
    }

    // ---- epilogue: out = alpha[o] * acc ----
#pragma unroll
    for (int ns = 0; ns < 4; ++ns) {
        const int o = n0 + wn * 64 + ns * 16 + lr;
        const float al = alpha[o];
#pragma unroll
        for (int ms = 0; ms < 2; ++ms) {
            const int mbase = m0 + wm * 32 + ms * 16 + laneHalf * 8;
            v8f a = acc[ms][ns];
#pragma unroll
            for (int v = 0; v < 8; ++v)
                out[(size_t)(mbase + v) * OUT_F + o] = al * a[v];
        }
    }
}

// ---------------------------------------------------------------------------
extern "C" void kernel_launch(void* const* d_in, const int* in_sizes, int n_in,
                              void* d_out, int out_size, void* d_ws, size_t ws_size,
                              hipStream_t stream) {
    const float* x      = (const float*)d_in[0];
    const int*   packed = (const int*)d_in[1];
    const float* alpha  = (const float*)d_in[2];
    const float* mu     = (const float*)d_in[3];
    const float* bias   = (const float*)d_in[4];
    const float* lora_A = (const float*)d_in[5];
    const float* lora_B = (const float*)d_in[6];
    float* out = (float*)d_out;

    // workspace: Aaug bf16 [256][4160] (2,129,920 B) + Baux bf16 [11008][64]
    unsigned short* Aaug = (unsigned short*)d_ws;
    unsigned short* Baux = (unsigned short*)((char*)d_ws + (size_t)MROWS * KAUG * 2);

    prep_a_kernel<<<MROWS, 256, 0, stream>>>(x, lora_A, Aaug);
    prep_baux_kernel<<<(OUT_F + 255) / 256, 256, 0, stream>>>(alpha, mu, bias, lora_B, Baux);

    dim3 grid(OUT_F / BN, MROWS / BM);   // 86 x 2
    ternary_gemm_kernel<<<grid, 256, 0, stream>>>(Aaug, Baux, packed, alpha, out);
}
